// Decoder_44607530336550
// MI455X (gfx1250) — compile-verified, matrix operand order checked by reading, not run on hardware
//
#include <hip/hip_runtime.h>
#include <math.h>

// ---------------------------------------------------------------------------
// Decoder recurrence for MI455X (gfx1250, wave32, WMMA).
// One wave = 16 batch rows; 16 blocks x 32 threads cover B=256.
// All GEMMs -> v_wmma_f32_16x16x32_f16, weights prepacked to B-fragment layout.
// ---------------------------------------------------------------------------

typedef __attribute__((ext_vector_type(16))) _Float16 v16h;
typedef __attribute__((ext_vector_type(8)))  _Float16 v8h;
typedef __attribute__((ext_vector_type(8)))  float    v8f;

static constexpr int B_ = 256;
static constexpr int S_ = 8;
static constexpr int C_ = 8;
static constexpr int H_ = 56;
static constexpr int L_ = 501;

// packed-weight fragment offsets (halves); each fragment = 32 lanes * 16 halves = 512
static constexpr int OFF_LIN1 = 0;                       // KT=16 NT=4
static constexpr int OFF_AV1  = OFF_LIN1 + 16 * 4 * 512; // KT=2  NT=7
static constexpr int OFF_AV2  = OFF_AV1  +  2 * 7 * 512; // KT=4  NT=1
static constexpr int OFF_AE1  = OFF_AV2  +  4 * 1 * 512; // KT=4  NT=14
static constexpr int OFF_AE2  = OFF_AE1  +  4 *14 * 512; // KT=7  NT=1
static constexpr int OFF_GATE = OFF_AE2  +  7 * 1 * 512; // KT=2  NT=4
static constexpr int OFF_MAP  = OFF_GATE +  2 * 4 * 512; // KT=2  NT=4
static constexpr int OFF_GIH  = OFF_MAP  +  2 * 4 * 512; // KT=1  NT=11
static constexpr int OFF_GHH  = OFF_GIH  +  1 *11 * 512; // KT=2  NT=11

union V16U { v16h v; v8h h[2]; };

__device__ __forceinline__ float sigm(float x) { return 1.f / (1.f + __expf(-x)); }

// A-operand fragment from LDS f16 activations (16 rows x strideA halves).
// Per-lane half pattern (16-bit A 16x32 ISA layout):
//   lane<16 : k = kt*32 + {0..7, 16..23};  lane>=16 : k = kt*32 + {8..15, 24..31}
__device__ __forceinline__ v16h load_a_frag(const _Float16* act, int strideA, int kt, int lane) {
  const int m    = lane & 15;
  const int base = kt * 32 + ((lane >> 4) << 3);
  const _Float16* p = act + m * strideA + base;
  V16U u;
  u.h[0] = *(const v8h*)(p);
  u.h[1] = *(const v8h*)(p + 16);
  return u.v;
}

// B-operand fragment from prepacked global weights: [frag][lane][16 halves]
__device__ __forceinline__ v16h load_b_frag(const _Float16* wp, int frag, int lane) {
  const _Float16* p = wp + ((frag << 5) + lane) * 16;
  V16U u;
  u.h[0] = *(const v8h*)(p);
  u.h[1] = *(const v8h*)(p + 8);
  return u.v;
}

// D = A(16xKT*32) * B + bias, optional relu; result f32 in LDS (row stride strideO).
// D layout: vgpr r, lanes 0-15 -> (m=r, n=lane); lanes 16-31 -> (m=8+r, n=lane-16)
__device__ __forceinline__ void gemm_ld(const _Float16* act, int strideA,
                                        const _Float16* wp, int KT, int NT,
                                        const float* bias, int Nreal,
                                        float* out, int strideO, int relu, int lane) {
  const int n  = lane & 15;
  const int mb = (lane >> 4) << 3;
  for (int nt = 0; nt < NT; ++nt) {
    v8f acc = {};
    for (int kt = 0; kt < KT; ++kt) {
      v16h a = load_a_frag(act, strideA, kt, lane);
      v16h b = load_b_frag(wp, kt * NT + nt, lane);
      acc = __builtin_amdgcn_wmma_f32_16x16x32_f16(false, a, false, b, (short)0, acc, false, false);
    }
    const int gn = nt * 16 + n;
    if (gn < Nreal) {
      const float bv = bias[gn];
#pragma unroll
      for (int r = 0; r < 8; ++r) {
        float v = acc[r] + bv;
        if (relu) v = fmaxf(v, 0.f);
        out[(mb + r) * strideO + gn] = v;
      }
    }
  }
}

// f32 LDS rows (stride strideF, Kreal cols) -> zero-padded f16 LDS (16 x Kpad)
__device__ __forceinline__ void cvt_f32_to_h(const float* src, int strideF, int Kreal,
                                             int Kpad, _Float16* dst, int lane) {
  const int total = 16 * Kpad;
  for (int i = lane; i < total; i += 32) {
    const int m = i / Kpad, k = i - m * Kpad;
    dst[i] = (_Float16)((k < Kreal) ? src[m * strideF + k] : 0.f);
  }
}

// GRU cell: hv = (1-z)*tanh(i_n + r*h_n) + z*hid  (hid==nullptr means zeros)
__device__ __forceinline__ void gru_step(const float* ne, int b0, int index,
                                         const float* hid, float* hv,
                                         _Float16* actH, float* gi, float* gh,
                                         const _Float16* wp, const float* gib,
                                         const float* ghb, int lane) {
  // x (B x 8) -> A operand, Kpad=32
  for (int i = lane; i < 16 * 32; i += 32) {
    const int m = i >> 5, k = i & 31;
    actH[i] = (_Float16)((k < C_) ? ne[((size_t)(b0 + m) * S_ + index) * C_ + k] : 0.f);
  }
  gemm_ld(actH, 32, wp + OFF_GIH, 1, 11, gib, 3 * H_, gi, 168, 0, lane);
  // hid -> A operand, Kpad=64
  for (int i = lane; i < 16 * 64; i += 32) {
    const int m = i >> 6, k = i & 63;
    actH[i] = (_Float16)((hid && k < H_) ? hid[m * H_ + k] : 0.f);
  }
  gemm_ld(actH, 64, wp + OFF_GHH, 2, 11, ghb, 3 * H_, gh, 168, 0, lane);
  // pointwise
  for (int i = lane; i < 16 * H_; i += 32) {
    const int m = i / H_, n = i - m * H_;
    const float ir = gi[m * 168 + n], iz = gi[m * 168 + H_ + n], in_ = gi[m * 168 + 2 * H_ + n];
    const float hr = gh[m * 168 + n], hz = gh[m * 168 + H_ + n], hn = gh[m * 168 + 2 * H_ + n];
    const float r  = sigm(ir + hr);
    const float zz = sigm(iz + hz);
    const float nn = tanhf(in_ + r * hn);
    const float hd = hid ? hid[m * H_ + n] : 0.f;
    hv[m * H_ + n] = (1.f - zz) * nn + zz * hd;
  }
}

// ---------------------------------------------------------------------------
// Weight pre-pack: f32 (Nreal x Kreal, row-major) -> f16 B fragments
// ---------------------------------------------------------------------------
extern "C" __global__ void pack_weight(const float* __restrict__ W, _Float16* __restrict__ dst,
                                       int Nreal, int Kreal, int NT, int KT) {
  const int idx = blockIdx.x * 256 + threadIdx.x;
  const int total = KT * NT * 512;
  if (idx >= total) return;
  const int frag   = idx >> 9;
  const int within = idx & 511;
  const int lane   = within >> 4;
  const int h      = within & 15;
  const int nt = frag % NT;
  const int kt = frag / NT;
  const int k = kt * 32 + ((h < 8) ? h : h + 8) + ((lane >= 16) ? 8 : 0);
  const int n = nt * 16 + (lane & 15);
  const float v = (n < Nreal && k < Kreal) ? W[(size_t)n * Kreal + k] : 0.f;
  dst[idx] = (_Float16)v;
}

// ---------------------------------------------------------------------------
// Main recurrence kernel: one wave per 16-batch tile
// ---------------------------------------------------------------------------
extern "C" __global__ void __launch_bounds__(32)
decoder_main(const float* __restrict__ z, const float* __restrict__ dep,
             const float* __restrict__ ne, const _Float16* __restrict__ wp,
             const float* __restrict__ lin1_b, const float* __restrict__ av_b1,
             const float* __restrict__ av_b2, const float* __restrict__ ae_b1,
             const float* __restrict__ ae_b2, const float* __restrict__ gate_b,
             const float* __restrict__ map_b, const float* __restrict__ gib,
             const float* __restrict__ ghb,
             float* __restrict__ out_dep, float* __restrict__ out_enc) {
  __shared__ __align__(16) _Float16 actH[16 * 256];     // GEMM A staging (f16)
  __shared__ __align__(16) _Float16 nhsH[16 * 8 * 56];  // node-hidden states (f16)
  __shared__ float hv[16 * 56];                         // carry / graph_state
  __shared__ float hin[16 * 56];
  __shared__ float gi[16 * 168];
  __shared__ float tA[16 * 224];                        // relu hidden / gh
  __shared__ float tB[16 * 56];
  __shared__ float tC[16 * 56];
  __shared__ float t16[16 * 16];
  __shared__ float cv[56];                              // sigmoid(gate_b)*map_b
  __shared__ float depL[16 * 8];

  const int lane = threadIdx.x;
  const int b0   = blockIdx.x * 16;

  // ---- init state ----
  for (int i = lane; i < 16 * 8 * H_; i += 32) nhsH[i] = (_Float16)0.f;
  for (int i = lane; i < H_; i += 32) cv[i] = sigm(gate_b[i]) * map_b[i];

  // ---- graph_state0 = z @ lin1_w^T + lin1_b  (K=501 split into 2 LDS passes) ----
  {
    v8f acc[4] = {{}, {}, {}, {}};
    for (int half = 0; half < 2; ++half) {
      for (int i = lane; i < 16 * 256; i += 32) {
        const int m = i >> 8, k = i & 255;
        const int gk = (half << 8) + k;
        actH[i] = (_Float16)((gk < L_) ? z[(size_t)(b0 + m) * L_ + gk] : 0.f);
      }
      for (int nt = 0; nt < 4; ++nt)
        for (int kt = 0; kt < 8; ++kt) {
          v16h a = load_a_frag(actH, 256, kt, lane);
          v16h b = load_b_frag(wp + OFF_LIN1, ((half * 8 + kt) << 2) + nt, lane);
          acc[nt] = __builtin_amdgcn_wmma_f32_16x16x32_f16(false, a, false, b, (short)0,
                                                           acc[nt], false, false);
        }
    }
    const int n = lane & 15, mb = (lane >> 4) << 3;
    for (int nt = 0; nt < 4; ++nt) {
      const int gn = nt * 16 + n;
      if (gn < H_) {
        const float bv = lin1_b[gn];
#pragma unroll
        for (int r = 0; r < 8; ++r) hv[(mb + r) * H_ + gn] = acc[nt][r] + bv;
      }
    }
  }

  // ---- outer scan over node index ----
  for (int index = 0; index < S_; ++index) {
    // stage dep_graph row for this index
    for (int i = lane; i < 16 * 8; i += 32) {
      const int m = i >> 3, s = i & 7;
      depL[i] = dep[((size_t)(b0 + m) * S_ + index) * S_ + s];
    }

    // ---- logits = softmax(relu(gs@av1^T+b1)@av2^T+b2) ----
    cvt_f32_to_h(hv, H_, H_, 64, actH, lane);
    gemm_ld(actH, 64, wp + OFF_AV1, 2, 7, av_b1, 2 * H_, tA, 112, 1, lane);
    cvt_f32_to_h(tA, 112, 112, 128, actH, lane);
    gemm_ld(actH, 128, wp + OFF_AV2, 4, 1, av_b2, C_, t16, 16, 0, lane);
    if (lane < 16) {
      const int m = lane;
      float mx = -1e30f;
      for (int c = 0; c < C_; ++c) mx = fmaxf(mx, t16[m * 16 + c]);
      float e[C_], sum = 0.f;
      for (int c = 0; c < C_; ++c) { e[c] = __expf(t16[m * 16 + c] - mx); sum += e[c]; }
      const float inv = 1.f / sum;
      for (int c = 0; c < C_; ++c)
        out_enc[((size_t)(b0 + m) * S_ + index) * C_ + c] = e[c] * inv;
    }

    // ---- hv0 = gru(x, index==0 ? graph_state : 0)  (in-place hid==hv is safe) ----
    gru_step(ne, b0, index, (index == 0) ? hv : nullptr, hv, actH, gi, tA, wp, gib, ghb, lane);

    // inactive edge slots are exact zeros in the reference
    if (lane < 16)
      for (int j = index; j < S_; ++j)
        out_dep[((size_t)(b0 + lane) * S_ + index) * S_ + j] = 0.f;

    // ---- inner scan j = index-1 .. 0 (only active iterations mutate state) ----
    for (int j = index - 1; j >= 0; --j) {
      // edge = ae2(relu(ae1([hv, nhs[:,j]])))
      for (int i = lane; i < 16 * 128; i += 32) {
        const int m = i >> 7, k = i & 127;
        float v = 0.f;
        if (k < H_)            v = hv[m * H_ + k];
        else if (k < 2 * H_)   v = (float)nhsH[(m * 8 + j) * H_ + (k - H_)];
        actH[i] = (_Float16)v;
      }
      gemm_ld(actH, 128, wp + OFF_AE1, 4, 14, ae_b1, 4 * H_, tA, 224, 1, lane);
      cvt_f32_to_h(tA, 224, 224, 224, actH, lane);
      gemm_ld(actH, 224, wp + OFF_AE2, 7, 1, ae_b2, 1, t16, 16, 0, lane);
      if (lane < 16)
        out_dep[((size_t)(b0 + lane) * S_ + index) * S_ + j] = t16[lane * 16];

      // h_in = sum_s sigmoid(a_s@gate^T+gb) * (a_s@map^T+mb),  a_s = nhs[:,s]*anc[:,s]
      // s<j rows are zero -> constant contribution j * sigmoid(gb)*mb (precomputed cv)
      for (int i = lane; i < 16 * H_; i += 32) hin[i] = (float)j * cv[i % H_];
      for (int s = j; s < S_; ++s) {
        for (int i = lane; i < 16 * 64; i += 32) {
          const int m = i >> 6, k = i & 63;
          const float v = (k < H_) ? (float)nhsH[(m * 8 + s) * H_ + k] * depL[m * 8 + s] : 0.f;
          actH[i] = (_Float16)v;
        }
        gemm_ld(actH, 64, wp + OFF_GATE, 2, 4, gate_b, H_, tB, H_, 0, lane);
        gemm_ld(actH, 64, wp + OFF_MAP,  2, 4, map_b,  H_, tC, H_, 0, lane);
        for (int i = lane; i < 16 * H_; i += 32) hin[i] += sigm(tB[i]) * tC[i];
      }

      // hv = gru(x, h_in); nhs[:,index] = hv
      gru_step(ne, b0, index, hin, hv, actH, gi, tA, wp, gib, ghb, lane);
      for (int i = lane; i < 16 * H_; i += 32) {
        const int m = i / H_, n = i - m * H_;
        nhsH[(m * 8 + index) * H_ + n] = (_Float16)hv[i];
      }
    }

    // nhs[:,index] = hv (covers index==0; redundant-but-cheap otherwise)
    for (int i = lane; i < 16 * H_; i += 32) {
      const int m = i / H_, n = i - m * H_;
      nhsH[(m * 8 + index) * H_ + n] = (_Float16)hv[i];
    }
  }
}

// ---------------------------------------------------------------------------
// Host launcher
// ---------------------------------------------------------------------------
static void pack(const float* W, _Float16* dst, int Nreal, int Kreal, int KT, int NT,
                 hipStream_t s) {
  const int total = KT * NT * 512;
  pack_weight<<<(total + 255) / 256, 256, 0, s>>>(W, dst, Nreal, Kreal, NT, KT);
}

extern "C" void kernel_launch(void* const* d_in, const int* in_sizes, int n_in,
                              void* d_out, int out_size, void* d_ws, size_t ws_size,
                              hipStream_t stream) {
  (void)in_sizes; (void)n_in; (void)out_size; (void)ws_size;
  const float* z        = (const float*)d_in[0];
  const float* dep      = (const float*)d_in[1];
  const float* ne       = (const float*)d_in[2];
  const float* lin1_w   = (const float*)d_in[3];
  const float* lin1_b   = (const float*)d_in[4];
  const float* av_w1    = (const float*)d_in[5];
  const float* av_b1    = (const float*)d_in[6];
  const float* av_w2    = (const float*)d_in[7];
  const float* av_b2    = (const float*)d_in[8];
  const float* ae_w1    = (const float*)d_in[9];
  const float* ae_b1    = (const float*)d_in[10];
  const float* ae_w2    = (const float*)d_in[11];
  const float* ae_b2    = (const float*)d_in[12];
  const float* gate_w   = (const float*)d_in[13];
  const float* gate_b   = (const float*)d_in[14];
  const float* map_w    = (const float*)d_in[15];
  const float* map_b    = (const float*)d_in[16];
  const float* gru_w_ih = (const float*)d_in[17];
  const float* gru_w_hh = (const float*)d_in[18];
  const float* gru_b_ih = (const float*)d_in[19];
  const float* gru_b_hh = (const float*)d_in[20];

  _Float16* wp = (_Float16*)d_ws;  // ~200 KB of packed f16 fragments

  pack(lin1_w,   wp + OFF_LIN1, H_,     L_,     16, 4,  stream);
  pack(av_w1,    wp + OFF_AV1,  2 * H_, H_,     2,  7,  stream);
  pack(av_w2,    wp + OFF_AV2,  C_,     2 * H_, 4,  1,  stream);
  pack(ae_w1,    wp + OFF_AE1,  4 * H_, 2 * H_, 4,  14, stream);
  pack(ae_w2,    wp + OFF_AE2,  1,      4 * H_, 7,  1,  stream);
  pack(gate_w,   wp + OFF_GATE, H_,     H_,     2,  4,  stream);
  pack(map_w,    wp + OFF_MAP,  H_,     H_,     2,  4,  stream);
  pack(gru_w_ih, wp + OFF_GIH,  3 * H_, C_,     1,  11, stream);
  pack(gru_w_hh, wp + OFF_GHH,  3 * H_, H_,     2,  11, stream);

  float* out_dep = (float*)d_out;
  float* out_enc = out_dep + (size_t)B_ * S_ * S_;

  decoder_main<<<dim3(B_ / 16), dim3(32), 0, stream>>>(
      z, dep, ne, wp, lin1_b, av_b1, av_b2, ae_b1, ae_b2, gate_b, map_b,
      gru_b_ih, gru_b_hh, out_dep, out_enc);
}
